// SimpleRNN_58609123721294
// MI455X (gfx1250) — compile-verified, matrix operand order checked by reading, not run on hardware
//
#include <hip/hip_runtime.h>
#include <math.h>

#define BATCH 128
#define SEQ   1024
#define INF_  256
#define HID   512
#define OUTF  256
#define NWG_SCAN 64
#define LDS_PITCH 520   // 512 elems + 8-elem (16B) pad per row -> conflict-free b128 reads

typedef __attribute__((ext_vector_type(16))) __bf16 v16bf;
typedef __attribute__((ext_vector_type(8)))  float  v8f;
typedef unsigned int u32x4 __attribute__((ext_vector_type(4)));
typedef int          i32x8 __attribute__((ext_vector_type(8)));
typedef int          i32x4 __attribute__((ext_vector_type(4)));

#if __has_builtin(__builtin_amdgcn_tensor_load_to_lds) && __has_builtin(__builtin_amdgcn_s_wait_tensorcnt)
#define USE_TDM 1
#else
#define USE_TDM 0
#endif

union Frag16 { v16bf v; unsigned short s[16]; unsigned d[8]; uint4 q[2]; };
union FragC  { v8f   v; float f[8]; };

__device__ __forceinline__ unsigned short f32_to_bf16_rne(float x) {
    unsigned u = __float_as_uint(x);
    unsigned r = u + 0x7FFFu + ((u >> 16) & 1u);
    return (unsigned short)(r >> 16);
}
__device__ __forceinline__ float bf16_bits_to_f32(unsigned short s) {
    return __uint_as_float(((unsigned)s) << 16);
}
// Pack two f32 -> two bf16 (round-half-up) with a single v_perm_b32.
__device__ __forceinline__ unsigned pack2_bf16(float lo, float hi) {
    unsigned a = __float_as_uint(lo) + 0x8000u;
    unsigned b = __float_as_uint(hi) + 0x8000u;
    return __builtin_amdgcn_perm(b, a, 0x07060302u);   // {hi[31:16], lo[31:16]}
}

// A/B fragment for V_WMMA_F32_16X16X32_BF16 from a row-major bf16 matrix.
// lane<16: elems 0..7 -> K=k0+0..7, elems 8..15 -> K=k0+16..23; lane>=16: +8.
__device__ __forceinline__ v16bf load_frag_bf16(const unsigned short* base, long lda,
                                                long row0, int k0, int lane) {
    int half = (lane >> 4) & 1;
    long r = row0 + (lane & 15);
    const unsigned short* p = base + r * lda + (long)k0 + half * 8;
    Frag16 f;
    f.q[0] = *reinterpret_cast<const uint4*>(p);
    f.q[1] = *reinterpret_cast<const uint4*>(p + 16);
    return f.v;
}
// Same fragment from the LDS-staged tile (rows 0..15, pitch LDS_PITCH).
__device__ __forceinline__ v16bf load_frag_lds(const unsigned short* smem, int k0, int lane) {
    int half = (lane >> 4) & 1;
    int r = lane & 15;
    const unsigned short* p = smem + r * LDS_PITCH + k0 + half * 8;
    Frag16 f;
    f.q[0] = *reinterpret_cast<const uint4*>(p);
    f.q[1] = *reinterpret_cast<const uint4*>(p + 16);
    return f.v;
}
// Fragment sourced from fp32 memory, packed via v_perm.
__device__ __forceinline__ v16bf load_frag_f32(const float* base, long lda,
                                               long row0, int k0, int lane) {
    int half = (lane >> 4) & 1;
    long r = row0 + (lane & 15);
    const float* p = base + r * lda + (long)k0 + half * 8;
    Frag16 f;
#pragma unroll
    for (int e = 0; e < 4; ++e) {
        f.d[e]     = pack2_bf16(p[2 * e],      p[2 * e + 1]);
        f.d[4 + e] = pack2_bf16(p[16 + 2 * e], p[16 + 2 * e + 1]);
    }
    return f.v;
}

// ---------------- Phase 0: one-time bf16 weight staging ----------------
__global__ void __launch_bounds__(256) prep_weights_kernel(const float* __restrict__ W_i2h,
                                                           const float* __restrict__ W_h2o,
                                                           unsigned short* __restrict__ Wx,
                                                           unsigned short* __restrict__ Wh,
                                                           unsigned short* __restrict__ Wo) {
    const long stride = (long)gridDim.x * blockDim.x;
    const long tid = (long)blockIdx.x * blockDim.x + threadIdx.x;
    for (long i = tid; i < (long)HID * INF_; i += stride) {
        long j = i / INF_, k = i % INF_;
        Wx[i] = f32_to_bf16_rne(W_i2h[j * (INF_ + HID) + k]);
    }
    for (long i = tid; i < (long)HID * HID; i += stride) {
        long j = i / HID, k = i % HID;
        Wh[i] = f32_to_bf16_rne(W_i2h[j * (INF_ + HID) + INF_ + k]);
    }
    for (long i = tid; i < (long)OUTF * HID; i += stride) {
        Wo[i] = f32_to_bf16_rne(W_h2o[i]);
    }
}

// ---------------- Phase 1: xW = x @ Wx^T + b_i2h (bf16 out) ----------------
__global__ void __launch_bounds__(128) xw_gemm_kernel(const float* __restrict__ x,
                                                      const unsigned short* __restrict__ Wx,
                                                      const float* __restrict__ b_i2h,
                                                      unsigned short* __restrict__ xw) {
    const int lane = threadIdx.x & 31;
    const int wave = threadIdx.x >> 5;
    const long grp = (long)blockIdx.x * 4 + wave;   // M/16 * HID/64 groups
    const int  nG  = HID / 64;                      // 8
    const long m0  = (grp / nG) * 16;
    const int  n0  = (int)(grp % nG) * 64;

    v8f c[4] = {};
#pragma unroll
    for (int kc = 0; kc < INF_ / 32; ++kc) {
        v16bf a = load_frag_f32(x, INF_, m0, kc * 32, lane);
#pragma unroll
        for (int j = 0; j < 4; ++j) {
            v16bf b = load_frag_bf16(Wx, INF_, n0 + j * 16, kc * 32, lane);
            c[j] = __builtin_amdgcn_wmma_f32_16x16x32_bf16(false, a, false, b, (short)0, c[j], false, false);
        }
    }
    const int half = lane >> 4;
#pragma unroll
    for (int j = 0; j < 4; ++j) {
        const int n = n0 + j * 16 + (lane & 15);
        const float bias = b_i2h[n];
        FragC fc; fc.v = c[j];
#pragma unroll
        for (int r = 0; r < 8; ++r) {
            long m = m0 + half * 8 + r;
            xw[m * HID + n] = f32_to_bf16_rne(fc.f[r] + bias);
        }
    }
}

// ---------------- Phase 2: persistent recurrent scan ----------------
// 64 WGs x 128 threads = 256 waves = 256 (16x16) tiles of h (8 batch x 32 hid).
// The 4 waves of a WG share one batch block: h_{t-1} tile is staged once per
// WG per step into LDS by the Tensor Data Mover (padded to avoid bank
// conflicts), then consumed via ds_load_b128. Wh slice persistent in VGPRs.
__global__ void __launch_bounds__(128) rnn_scan_kernel(const unsigned short* __restrict__ Wh,
                                                       const unsigned short* __restrict__ xw,
                                                       unsigned short* __restrict__ hs,
                                                       float* __restrict__ h_last,
                                                       unsigned* __restrict__ barrier_cnt) {
    __shared__ unsigned short smem[16 * LDS_PITCH];   // 16.6 KB staged A tile

    const int lane  = threadIdx.x & 31;
    const int wave  = threadIdx.x >> 5;
    const int wid   = blockIdx.x * 4 + wave;   // 0..255
    const long b0   = (long)(wid >> 5) * 16;   // batch block (same for all 4 waves of WG)
    const int  j0   = (wid & 31) * 16;         // hidden block
    const int  half = lane >> 4;
    const int  n    = j0 + (lane & 15);

    // Persistent B fragments: Wh rows j0..j0+15, full K = 0..511 (128 VGPRs).
    v16bf Bh[HID / 32];
#pragma unroll
    for (int kc = 0; kc < HID / 32; ++kc)
        Bh[kc] = load_frag_bf16(Wh, HID, j0, kc * 32, lane);

    const long ldh = (long)SEQ * HID;          // batch-row stride in hs[b][t][j]
#if USE_TDM
    const unsigned lds_base = (unsigned)(uintptr_t)(&smem[0]);   // low 32 bits = LDS offset
#endif

    for (int t = 0; t < SEQ; ++t) {
        v8f c0 = {}, c1 = {};
        if (t > 0) {
#if USE_TDM
            if (wave == 0) {
                // D# group 0: count=1, lds_addr, 57-bit global tile addr, type=2
                const unsigned long long ga =
                    (unsigned long long)(uintptr_t)(hs + b0 * ldh + (long)(t - 1) * HID);
                u32x4 g0;
                g0.x = 1u;
                g0.y = lds_base;
                g0.z = (unsigned)ga;
                g0.w = (unsigned)(ga >> 32) | (2u << 30);
                // D# group 1: data_size=2B; pad 4 dwords every 1KB (row pitch 1040B);
                // tensor 512 x 16, tile 512 x 16, dim0 stride = SEQ*HID elements.
                i32x8 g1;
                g1[0] = (int)((1u << 16) | (1u << 20) | (7u << 22) | (3u << 25));
                g1[1] = (int)(512u << 16);     // tensor_dim0 = 512
                g1[2] = (int)(16u << 16);      // tensor_dim1 = 16
                g1[3] = (int)(512u << 16);     // tile_dim0 = 512
                g1[4] = 16;                    // tile_dim1 = 16, tile_dim2 = 0
                g1[5] = (int)((unsigned)(SEQ * HID));   // tensor_dim0_stride lo32
                g1[6] = 0;
                g1[7] = 0;
                i32x4 gz4 = {0, 0, 0, 0};
                i32x8 gz8 = {0, 0, 0, 0, 0, 0, 0, 0};
                __builtin_amdgcn_tensor_load_to_lds(g0, g1, gz4, gz4, gz8, 0);
                __builtin_amdgcn_s_wait_tensorcnt(0);
            }
            __syncthreads();
#else
            // Fallback: cooperative global->LDS staging (with same padded pitch).
            const unsigned short* hprev = hs + b0 * ldh + (long)(t - 1) * HID;
            for (int idx = threadIdx.x; idx < 16 * 64; idx += 128) {
                int row = idx >> 6, chunk = idx & 63;
                *reinterpret_cast<uint4*>(&smem[row * LDS_PITCH + chunk * 8]) =
                    *reinterpret_cast<const uint4*>(&hprev[(long)row * ldh + chunk * 8]);
            }
            __syncthreads();
#endif
#pragma unroll
            for (int kc = 0; kc < HID / 32; kc += 2) {
                v16bf a0 = load_frag_lds(smem, kc * 32, lane);
                v16bf a1 = load_frag_lds(smem, (kc + 1) * 32, lane);
                c0 = __builtin_amdgcn_wmma_f32_16x16x32_bf16(false, a0, false, Bh[kc],     (short)0, c0, false, false);
                c1 = __builtin_amdgcn_wmma_f32_16x16x32_bf16(false, a1, false, Bh[kc + 1], (short)0, c1, false, false);
            }
        }
        FragC f0, f1; f0.v = c0; f1.v = c1;
#pragma unroll
        for (int r = 0; r < 8; ++r) {
            long b = b0 + half * 8 + r;
            float pre = f0.f[r] + f1.f[r] + bf16_bits_to_f32(xw[(b * SEQ + t) * HID + n]);
            float h = tanhf(pre);
            hs[b * ldh + (long)t * HID + n] = f32_to_bf16_rne(h);
            if (t == SEQ - 1) h_last[b * HID + n] = h;
            if (t + 1 < SEQ) __builtin_prefetch(&xw[(b * SEQ + t + 1) * HID + n], 0, 1);
        }
        // ---- grid-wide barrier between timesteps ----
        __threadfence();
        __syncthreads();
        if (threadIdx.x == 0) {
            __hip_atomic_fetch_add(barrier_cnt, 1u, __ATOMIC_ACQ_REL, __HIP_MEMORY_SCOPE_AGENT);
            const unsigned target = (unsigned)NWG_SCAN * (unsigned)(t + 1);
            while (__hip_atomic_load(barrier_cnt, __ATOMIC_ACQUIRE, __HIP_MEMORY_SCOPE_AGENT) < target)
                __builtin_amdgcn_s_sleep(1);
        }
        __syncthreads();
    }
}

// ---------------- Phase 3: out = hs @ Wo^T + b_h2o (fp32 out) ----------------
__global__ void __launch_bounds__(128) out_gemm_kernel(const unsigned short* __restrict__ hs,
                                                       const unsigned short* __restrict__ Wo,
                                                       const float* __restrict__ b_h2o,
                                                       float* __restrict__ out) {
    const int lane = threadIdx.x & 31;
    const int wave = threadIdx.x >> 5;
    const long grp = (long)blockIdx.x * 4 + wave;   // M/16 * OUTF/64 groups
    const int  nG  = OUTF / 64;                     // 4
    const long m0  = (grp / nG) * 16;
    const int  n0  = (int)(grp % nG) * 64;

    v8f c[4] = {};
#pragma unroll 2
    for (int kc = 0; kc < HID / 32; ++kc) {
        v16bf a = load_frag_bf16(hs, HID, m0, kc * 32, lane);
#pragma unroll
        for (int j = 0; j < 4; ++j) {
            v16bf b = load_frag_bf16(Wo, HID, n0 + j * 16, kc * 32, lane);
            c[j] = __builtin_amdgcn_wmma_f32_16x16x32_bf16(false, a, false, b, (short)0, c[j], false, false);
        }
    }
    const int half = lane >> 4;
#pragma unroll
    for (int j = 0; j < 4; ++j) {
        const int n = n0 + j * 16 + (lane & 15);
        const float bias = b_h2o[n];
        FragC fc; fc.v = c[j];
#pragma unroll
        for (int r = 0; r < 8; ++r) {
            long m = m0 + half * 8 + r;
            out[m * OUTF + n] = fc.f[r] + bias;
        }
    }
}

__global__ void init_barrier_kernel(unsigned* cnt) {
    if (threadIdx.x == 0 && blockIdx.x == 0) *cnt = 0u;
}

extern "C" void kernel_launch(void* const* d_in, const int* in_sizes, int n_in,
                              void* d_out, int out_size, void* d_ws, size_t ws_size,
                              hipStream_t stream) {
    const float* x     = (const float*)d_in[0];
    const float* W_i2h = (const float*)d_in[1];
    const float* b_i2h = (const float*)d_in[2];
    const float* W_h2o = (const float*)d_in[3];
    const float* b_h2o = (const float*)d_in[4];

    float* out    = (float*)d_out;
    float* h_last = out + (long)BATCH * SEQ * OUTF;

    unsigned short* xw = (unsigned short*)d_ws;                    // 128 MB
    unsigned short* hs = xw + (long)BATCH * SEQ * HID;             // 128 MB
    unsigned short* Wx = hs + (long)BATCH * SEQ * HID;             // 256 KB
    unsigned short* Wh = Wx + (long)HID * INF_;                    // 512 KB
    unsigned short* Wo = Wh + (long)HID * HID;                     // 256 KB
    unsigned* cnt      = (unsigned*)(Wo + (long)OUTF * HID);

    init_barrier_kernel<<<1, 32, 0, stream>>>(cnt);
    prep_weights_kernel<<<256, 256, 0, stream>>>(W_i2h, W_h2o, Wx, Wh, Wo);

    const long xwGrps = ((long)BATCH * SEQ / 16) * (HID / 64);     // 65536 waves
    xw_gemm_kernel<<<dim3((unsigned)(xwGrps / 4)), 128, 0, stream>>>(x, Wx, b_i2h, xw);

    rnn_scan_kernel<<<NWG_SCAN, 128, 0, stream>>>(Wh, xw, hs, h_last, cnt);

    const long outGrps = ((long)BATCH * SEQ / 16) * (OUTF / 64);   // 32768 waves
    out_gemm_kernel<<<dim3((unsigned)(outGrps / 4)), 128, 0, stream>>>(hs, Wo, b_h2o, out);
}
